// GCN_FusionLayer_89043261980847
// MI455X (gfx1250) — matrix operand order repaired and checked by reading, not compiled, not run on hardware
//
#include <hip/hip_runtime.h>

// ---------------------------------------------------------------------------
// GCN fusion layer for MI455X (gfx1250, wave32, WMMA bf16).
// adj (512MB) streamed from HBM exactly twice (deg + bmm) -> ~46us floor.
// B-fragments staged in LDS (shared by all 8 waves), A/B prefetched one
// k-chunk ahead in registers so WMMAs never wait on fresh loads.
// ---------------------------------------------------------------------------

typedef __attribute__((ext_vector_type(16))) __bf16 v16bf;
typedef __attribute__((ext_vector_type(8)))  float  v8f;

#define NEG_SLOPE 0.01f

constexpr int B = 8, N = 4096, IN = 256, OUT = 128;
constexpr int BSTR = 40;   // LDS row stride (bf16): 80B rows -> conflict-free frag reads

// workspace layout (bytes)
constexpr size_t OFF_TQ   = 0;                                // B*OUT f32
constexpr size_t OFF_WGVT = 4096;                             // OUT*IN bf16 (transposed)
constexpr size_t OFF_WGKT = OFF_WGVT + (size_t)OUT*IN*2;
constexpr size_t OFF_GATE = OFF_WGKT + (size_t)OUT*IN*2;      // B*N f32 (softmax+1)
constexpr size_t OFF_DIS  = OFF_GATE + (size_t)B*N*4;         // B*N f32 rsqrt(deg)
constexpr size_t OFF_GX   = OFF_DIS  + (size_t)B*N*4;         // B*N*OUT f32
constexpr size_t OFF_GK   = OFF_GX   + (size_t)B*N*OUT*4;     // B*N*OUT f32
constexpr size_t OFF_YPT  = OFF_GK   + (size_t)B*N*OUT*4;     // B*OUT*N bf16 (transposed)

__device__ inline v8f wmma_bf16(v16bf a, v16bf b, v8f c) {
  return __builtin_amdgcn_wmma_f32_16x16x32_bf16(
      false, a, false, b, (short)0, c, false, false);
}

// A-fragment element order (ISA 7.12.2, 16-bit A 16x32): per lane-half (lhi),
// elems 0..7 = K(kh*8..kh*8+7), elems 8..15 = K(16+kh*8..16+kh*8+7).
__device__ inline v16bf cvt16(float4 a, float4 b, float4 c, float4 d) {
  v16bf r;
  r[0]=(__bf16)a.x;  r[1]=(__bf16)a.y;  r[2]=(__bf16)a.z;  r[3]=(__bf16)a.w;
  r[4]=(__bf16)b.x;  r[5]=(__bf16)b.y;  r[6]=(__bf16)b.z;  r[7]=(__bf16)b.w;
  r[8]=(__bf16)c.x;  r[9]=(__bf16)c.y;  r[10]=(__bf16)c.z; r[11]=(__bf16)c.w;
  r[12]=(__bf16)d.x; r[13]=(__bf16)d.y; r[14]=(__bf16)d.z; r[15]=(__bf16)d.w;
  return r;
}

// Read a 16-bf16 B-fragment from LDS as two aligned ds_read_b128.
__device__ inline v16bf ldsfrag(const __bf16* p) {
  union { struct { uint4 lo, hi; } q; v16bf v; } u;
  u.q.lo = *(const uint4*)(p);
  u.q.hi = *(const uint4*)(p + 16);
  return u.v;
}

// ---- K0a: pack Wgv/Wgk transposed to bf16 (WT[h][i]) -----------------------
__global__ __launch_bounds__(256)
void pack_weights(const float* __restrict__ wgv, const float* __restrict__ wgk,
                  __bf16* __restrict__ wgvT, __bf16* __restrict__ wgkT) {
  int t = blockIdx.x * 256 + threadIdx.x;        // 65536 threads
  const float* W = (t < OUT * IN) ? wgv : wgk;
  __bf16* WT     = (t < OUT * IN) ? wgvT : wgkT;
  int e = t & (OUT * IN - 1);
  int h = e >> 8, i = e & 255;
  WT[h * IN + i] = (__bf16)W[i * OUT + h];
}

// ---- K0b: tabular path: T_q (to ws) and T_X (output part 1) ----------------
__global__ __launch_bounds__(128)
void tabular_kernel(const float* __restrict__ tab,
                    const float* __restrict__ wtq, const float* __restrict__ btq,
                    const float* __restrict__ wtv, const float* __restrict__ btv,
                    float* __restrict__ Tq, float* __restrict__ outT) {
  int b = blockIdx.x, h = threadIdx.x;
  const float* x = tab + b * IN;
  float sq = btq[h], sv = btv[h];
  for (int i = 0; i < IN; ++i) {
    float xv = x[i];
    sq += xv * wtq[i * OUT + h];
    sv += xv * wtv[i * OUT + h];
  }
  Tq[b * OUT + h]   = sq;
  outT[b * OUT + h] = (sv >= 0.f) ? sv : NEG_SLOPE * sv;
}

// ---- K1: dual node GEMM: G_X = lrelu(nf@Wgv+bgv), G_k = nf@Wgk+bgk ---------
// Block = 64 rows; waves 0-3 -> Wgv GEMM, waves 4-7 -> Wgk GEMM (same rows).
// Weight chunks staged in LDS; A prefetched one chunk ahead in registers.
__global__ __launch_bounds__(256)
void node_dual_gemm(const float* __restrict__ nf,
                    const __bf16* __restrict__ wgvT, const __bf16* __restrict__ wgkT,
                    const float* __restrict__ bgv, const float* __restrict__ bgk,
                    float* __restrict__ GX, float* __restrict__ GK) {
  __shared__ __bf16 wsh[2][128 * BSTR];      // [matrix][h*BSTR + k] : 20KB
  const int tid = threadIdx.x;
  const int wave = tid >> 5, lane = tid & 31;
  const int mat = wave >> 2;                 // 0 = Wgv path, 1 = Wgk path
  const int rbase = blockIdx.x * 64 + (wave & 3) * 16;
  const int lmod = lane & 15, lhi = lane >> 4;

  v8f acc[8];
  const v8f z = {0.f, 0.f, 0.f, 0.f, 0.f, 0.f, 0.f, 0.f};
#pragma unroll
  for (int t = 0; t < 8; ++t) acc[t] = z;

  const float* arow = nf + (size_t)(rbase + lmod) * IN;
  // staging role: threads 0-127 stage Wgv chunk, 128-255 stage Wgk chunk
  const int sm = tid >> 7, st = tid & 127;
  const __bf16* ssrc = (sm ? wgkT : wgvT) + (size_t)st * IN;
  const int sdst = st * BSTR;

  // prologue: stage chunk 0, load A chunk 0
  const float* ap = arow + lhi * 8;
  float4 f0 = *(const float4*)(ap);
  float4 f1 = *(const float4*)(ap + 4);
  float4 f2 = *(const float4*)(ap + 16);
  float4 f3 = *(const float4*)(ap + 20);
  uint4 s0 = *(const uint4*)(ssrc);
  uint4 s1 = *(const uint4*)(ssrc + 16);
  uint4 s2 = *(const uint4*)(ssrc + 8);
  uint4 s3 = *(const uint4*)(ssrc + 24);
  *(uint4*)&wsh[sm][sdst]      = s0;
  *(uint4*)&wsh[sm][sdst + 16] = s1;
  *(uint4*)&wsh[sm][sdst + 8]  = s2;
  *(uint4*)&wsh[sm][sdst + 24] = s3;
  __syncthreads();

  for (int i = 0; i < IN / 32; ++i) {
    int kn = (i + 1 < IN / 32) ? (i + 1) * 32 : 0;   // next chunk (wraps, harmless)
    const float* apn = arow + kn + lhi * 8;
    float4 g0 = *(const float4*)(apn);
    float4 g1 = *(const float4*)(apn + 4);
    float4 g2 = *(const float4*)(apn + 16);
    float4 g3 = *(const float4*)(apn + 20);
    uint4 t0 = *(const uint4*)(ssrc + kn);
    uint4 t1 = *(const uint4*)(ssrc + kn + 16);
    uint4 t2 = *(const uint4*)(ssrc + kn + 8);
    uint4 t3 = *(const uint4*)(ssrc + kn + 24);

    v16bf afrag = cvt16(f0, f1, f2, f3);
    v16bf bf[8];
#pragma unroll
    for (int t = 0; t < 8; ++t)
      bf[t] = ldsfrag(&wsh[mat][(t * 16 + lmod) * BSTR + lhi * 8]);
#pragma unroll
    for (int t = 0; t < 8; ++t) acc[t] = wmma_bf16(afrag, bf[t], acc[t]);

    __syncthreads();                         // all frag reads consumed
    *(uint4*)&wsh[sm][sdst]      = t0;
    *(uint4*)&wsh[sm][sdst + 16] = t1;
    *(uint4*)&wsh[sm][sdst + 8]  = t2;
    *(uint4*)&wsh[sm][sdst + 24] = t3;
    __syncthreads();                         // next chunk visible
    f0 = g0; f1 = g1; f2 = g2; f3 = g3;
  }

  const int col = lmod, ro = lhi * 8;        // C/D layout: row = v + 8*lhi
  const float* bias = mat ? bgk : bgv;
  float* dst = mat ? GK : GX;
#pragma unroll
  for (int t = 0; t < 8; ++t) {
    int h = t * 16 + col;
    float bb = bias[h];
#pragma unroll
    for (int v = 0; v < 8; ++v) {
      int r = rbase + ro + v;
      float x = acc[t][v] + bb;
      if (mat == 0) x = (x >= 0.f) ? x : NEG_SLOPE * x;
      dst[(size_t)r * OUT + h] = x;
    }
  }
}

// ---- K2: deg pass: dis[r] = rsqrt(1 + rowsum(adj[r])) ----------------------
__global__ __launch_bounds__(256)
void deg_kernel(const float* __restrict__ adj, float* __restrict__ dis) {
  int wave = threadIdx.x >> 5, lane = threadIdx.x & 31;
  int r = blockIdx.x * 8 + wave;             // global row in [0, B*N)
  const float4* row = (const float4*)(adj + (size_t)r * N);
  float s = 0.f;
  for (int it = 0; it < N / 128; ++it) {
    float4 v = row[it * 32 + lane];
    s += v.x + v.y + v.z + v.w;
  }
#pragma unroll
  for (int o = 16; o > 0; o >>= 1) s += __shfl_xor(s, o, 32);
  if (lane == 0) dis[r] = rsqrtf(s + 1.0f);
}

// ---- K3: attention logits + softmax + (att+1) gate, one block per batch ----
__global__ __launch_bounds__(1024)
void attn_gate(const float* __restrict__ Tq, const float* __restrict__ GK,
               float* __restrict__ gate) {
  __shared__ float logit[N];
  __shared__ float red[1024];
  int b = blockIdx.x, tid = threadIdx.x;
  int wave = tid >> 5, lane = tid & 31;
  float4 tq = ((const float4*)(Tq + b * OUT))[lane];
  for (int i = 0; i < N / 32; ++i) {
    int n = wave + i * 32;
    float4 g = ((const float4*)(GK + ((size_t)b * N + n) * OUT))[lane];
    float p = g.x * tq.x + g.y * tq.y + g.z * tq.z + g.w * tq.w;
#pragma unroll
    for (int o = 16; o > 0; o >>= 1) p += __shfl_xor(p, o, 32);
    if (lane == 0) logit[n] = p;
  }
  __syncthreads();
  float m = logit[tid];
  m = fmaxf(m, logit[tid + 1024]);
  m = fmaxf(m, logit[tid + 2048]);
  m = fmaxf(m, logit[tid + 3072]);
  red[tid] = m; __syncthreads();
  for (int s = 512; s > 0; s >>= 1) {
    if (tid < s) red[tid] = fmaxf(red[tid], red[tid + s]);
    __syncthreads();
  }
  float gmax = red[0]; __syncthreads();
  float s0 = __expf(logit[tid] - gmax) + __expf(logit[tid + 1024] - gmax) +
             __expf(logit[tid + 2048] - gmax) + __expf(logit[tid + 3072] - gmax);
  red[tid] = s0; __syncthreads();
  for (int s = 512; s > 0; s >>= 1) {
    if (tid < s) red[tid] += red[tid + s];
    __syncthreads();
  }
  float inv = 1.0f / red[0];
  for (int j = 0; j < 4; ++j) {
    int n = tid + j * 1024;
    gate[b * N + n] = __expf(logit[n] - gmax) * inv + 1.0f;
  }
}

// ---- K4: pack YpT[b][h][m] = bf16(dis*gate*G_X) via LDS transpose ----------
__global__ __launch_bounds__(256)
void pack_ypt(const float* __restrict__ GX, const float* __restrict__ gate,
              const float* __restrict__ dis, __bf16* __restrict__ ypT) {
  __shared__ __bf16 tile[128][130];          // stride 130 shorts -> conflict-free
  int b = blockIdx.y, m0 = blockIdx.x * 128, tid = threadIdx.x;
  for (int it = 0; it < 64; ++it) {
    int idx = it * 256 + tid;
    int h = idx & 127, mi = idx >> 7;
    int gr = b * N + m0 + mi;
    float s = dis[gr] * gate[gr];
    tile[h][mi] = (__bf16)(s * GX[(size_t)gr * OUT + h]);
  }
  __syncthreads();
  for (int it = 0; it < 64; ++it) {
    int idx = it * 256 + tid;
    int ml = idx & 127, h = idx >> 7;
    ypT[((size_t)b * OUT + h) * N + m0 + ml] = tile[h][ml];
  }
}

// ---- K5: fused GCN bmm: out = dis[n]*(adj @ Yp)[n,h] + dis^2*gate*G_X ------
// Block = 128 rows x 128 cols per batch slice; Yp chunk staged in LDS
// (double-buffered), A + B prefetched one chunk ahead in registers.
__global__ __launch_bounds__(256)
void gcn_bmm(const float* __restrict__ adj, const __bf16* __restrict__ ypT,
             const float* __restrict__ GX, const float* __restrict__ gate,
             const float* __restrict__ dis, float* __restrict__ outG) {
  __shared__ __bf16 bsh[2][128 * BSTR];      // 2 x 10KB double buffer
  const int b = blockIdx.y;
  const int tid = threadIdx.x;
  const int wave = tid >> 5, lane = tid & 31;
  const int rbase = blockIdx.x * 128 + wave * 16;
  const int lmod = lane & 15, lhi = lane >> 4;

  v8f acc[8];
  const v8f z = {0.f, 0.f, 0.f, 0.f, 0.f, 0.f, 0.f, 0.f};
#pragma unroll
  for (int t = 0; t < 8; ++t) acc[t] = z;

  const float* arow = adj + ((size_t)b * N + rbase + lmod) * N;
  // cooperative B staging: thread -> one 32B segment of the 32x128 chunk
  const int bh = tid >> 1, bhalf = tid & 1;
  const __bf16* bsrc = ypT + ((size_t)b * OUT + bh) * N + bhalf * 16;
  const int bdst = bh * BSTR + bhalf * 16;

  // prologue: chunk 0
  const float* ap = arow + lhi * 8;
  float4 f0 = *(const float4*)(ap);
  float4 f1 = *(const float4*)(ap + 4);
  float4 f2 = *(const float4*)(ap + 16);
  float4 f3 = *(const float4*)(ap + 20);
  uint4 bv = *(const uint4*)(bsrc);
  uint4 bv2 = *(const uint4*)(bsrc + 8);
  *(uint4*)&bsh[0][bdst] = bv;
  *(uint4*)&bsh[0][bdst + 8] = bv2;
  __syncthreads();

  for (int i = 0; i < N / 32; ++i) {
    int cur = i & 1, nxt = cur ^ 1;
    int mn = (i + 1 < N / 32) ? (i + 1) * 32 : 0;    // next chunk (wraps, harmless)
    // prefetch next chunk while computing current
    const float* apn = arow + mn + lhi * 8;
    float4 g0 = *(const float4*)(apn);
    float4 g1 = *(const float4*)(apn + 4);
    float4 g2 = *(const float4*)(apn + 16);
    float4 g3 = *(const float4*)(apn + 20);
    uint4 bn = *(const uint4*)(bsrc + mn);
    uint4 bn2 = *(const uint4*)(bsrc + mn + 8);

    v16bf afrag = cvt16(f0, f1, f2, f3);
    v16bf bf[8];
#pragma unroll
    for (int t = 0; t < 8; ++t)
      bf[t] = ldsfrag(&bsh[cur][(t * 16 + lmod) * BSTR + lhi * 8]);
#pragma unroll
    for (int t = 0; t < 8; ++t) acc[t] = wmma_bf16(afrag, bf[t], acc[t]);

    *(uint4*)&bsh[nxt][bdst] = bn;           // stage next into other buffer
    *(uint4*)&bsh[nxt][bdst + 8] = bn2;
    __syncthreads();
    f0 = g0; f1 = g1; f2 = g2; f3 = g3;
  }

  const int col = lmod, ro = lhi * 8;
#pragma unroll
  for (int v = 0; v < 8; ++v) {
    int r = rbase + ro + v;
    float dn = dis[b * N + r];
    float dscale = dn * dn * gate[b * N + r];  // diagonal (A+I) term
#pragma unroll
    for (int t = 0; t < 8; ++t) {
      int h = t * 16 + col;
      float gx = GX[((size_t)b * N + r) * OUT + h];
      outG[((size_t)b * N + r) * OUT + h] = dn * acc[t][v] + dscale * gx;
    }
  }
}

extern "C" void kernel_launch(void* const* d_in, const int* in_sizes, int n_in,
                              void* d_out, int out_size, void* d_ws, size_t ws_size,
                              hipStream_t stream) {
  (void)in_sizes; (void)n_in; (void)out_size; (void)ws_size;
  const float* tab = (const float*)d_in[0];
  const float* nf  = (const float*)d_in[1];
  const float* adj = (const float*)d_in[2];
  const float* wgv = (const float*)d_in[3];
  const float* bgv = (const float*)d_in[4];
  const float* wgk = (const float*)d_in[5];
  const float* bgk = (const float*)d_in[6];
  const float* wtq = (const float*)d_in[7];
  const float* btq = (const float*)d_in[8];
  const float* wtv = (const float*)d_in[9];
  const float* btv = (const float*)d_in[10];

  char* ws = (char*)d_ws;
  float*  Tq   = (float*)(ws + OFF_TQ);
  __bf16* wgvT = (__bf16*)(ws + OFF_WGVT);
  __bf16* wgkT = (__bf16*)(ws + OFF_WGKT);
  float*  gate = (float*)(ws + OFF_GATE);
  float*  dis  = (float*)(ws + OFF_DIS);
  float*  GX   = (float*)(ws + OFF_GX);
  float*  GK   = (float*)(ws + OFF_GK);
  __bf16* ypT  = (__bf16*)(ws + OFF_YPT);

  float* outT = (float*)d_out;           // T_X: B*OUT
  float* outG = (float*)d_out + B * OUT; // G_X: B*N*OUT

  pack_weights<<<256, 256, 0, stream>>>(wgv, wgk, wgvT, wgkT);
  tabular_kernel<<<B, OUT, 0, stream>>>(tab, wtq, btq, wtv, btv, Tq, outT);
  node_dual_gemm<<<(B * N) / 64, 256, 0, stream>>>(nf, wgvT, wgkT, bgv, bgk, GX, GK);
  deg_kernel<<<(B * N) / 8, 256, 0, stream>>>(adj, dis);
  attn_gate<<<B, 1024, 0, stream>>>(Tq, GK, gate);
  pack_ypt<<<dim3(N / 128, B), 256, 0, stream>>>(GX, gate, dis, ypT);
  gcn_bmm<<<dim3(N / 128, B), 256, 0, stream>>>(adj, ypT, GX, gate, dis, outG);
}
